// NXRODeepLearnableGCN_10127532884097
// MI455X (gfx1250) — matrix operand order repaired and checked by reading, not compiled, not run on hardware
//
#include <hip/hip_runtime.h>
#include <hip/hip_bf16.h>
#include <math.h>

// ---------------------------------------------------------------------------
// CDNA5 / gfx1250 fused Fourier-GCN.
//  * one wave32 per block; each block owns 16 consecutive samples
//  * dxdt  : batched WMMA GEMM  [16 x 320] x [320 x 64]  (B-frags precomputed
//            in fragment-native layout -> contiguous global b128 loads, L2-hot)
//  * layer1: per-sample 64x64x64 WMMA pair; W1^T kept in LDS in fragment-
//            native order (2x ds_load_b128 per frag); T stored transposed so
//            D-stores and GEMM2 B-frag loads are contiguous b128 ops
//  * relu+LN+W2 fused on GEMM2 accumulators via xor-shuffle row reductions
//  * no persistent VGPR fragment arrays -> live set ~100 VGPRs, no MSB mode
// ---------------------------------------------------------------------------

typedef __bf16 bf16;
typedef __attribute__((ext_vector_type(16))) __bf16 v16bf;
typedef __attribute__((ext_vector_type(8)))  float  v8f;

#define NB      5
#define NV      64
#define HID     64
#define BATCH   32768
#define LN_EPS  1e-5f
#define LDP     72                 // padded LDS row stride (bf16), 144B = 9*16B
#define XK      (NB * NV)          // 320 : fused (k,v) contraction dim
#define XLD     328                // padded Xk row stride, 656B = 41*16B
#define NKT     (XK / 32)          // 10 k-tiles for the dxdt GEMM
#define SPB     16                 // samples per block (one wave)
#define BLOCKS  (BATCH / SPB)      // 2048 blocks, uniform work

// ---- WMMA fragment helpers (contiguous k = 16*half + s order, wave32) --------------
// A-matrix 16x32 bf16, source row-major [m][k] with row stride ld
__device__ __forceinline__ v16bf load_a_frag(const bf16* S, int ld, int m0,
                                             int k0, int lane) {
  const int m  = m0 + (lane & 15);
  const int h8 = (lane >> 4) << 4;
  v16bf a;
#pragma unroll
  for (int g = 0; g < 8; ++g) {
    const int kb = k0 + (g << 1) + h8;
    a[2 * g]     = S[m * ld + kb];
    a[2 * g + 1] = S[m * ld + kb + 1];
  }
  return a;
}
// B-matrix 32x16 bf16 from a TRANSPOSED source T^T stored [n][k] row-major:
// lane n reads 16 contiguous bf16 -> two ds_load_b128
__device__ __forceinline__ v16bf load_b_frag_T(const bf16* S, int k0, int n0,
                                               int lane) {
  const bf16* p = S + (n0 + (lane & 15)) * LDP + k0 + ((lane >> 4) << 4);
  v16bf b;
#pragma unroll
  for (int i = 0; i < 16; ++i) b[i] = p[i];
  return b;
}
// fragment-native contiguous load (global L table / LDS W1 table)
__device__ __forceinline__ v16bf load_frag16(const bf16* p) {
  v16bf b;
#pragma unroll
  for (int i = 0; i < 16; ++i) b[i] = p[i];
  return b;
}

__device__ __forceinline__ v8f wmma_bf16(v16bf a, v16bf b, v8f c) {
  return __builtin_amdgcn_wmma_f32_16x16x32_bf16(false, a, false, b, (short)0, c,
                                                 false, false);
}

__device__ __forceinline__ float xor_reduce16(float v) {
  v += __shfl_xor(v, 1);
  v += __shfl_xor(v, 2);
  v += __shfl_xor(v, 4);
  v += __shfl_xor(v, 8);
  return v;
}

// ---- prep: A_norm (bf16), W1^T + L_basis in WMMA-B fragment-native layout --
__global__ __launch_bounds__(256) void gcn_prep(const float* __restrict__ A_param,
                                                const float* __restrict__ W1,
                                                const float* __restrict__ L_basis,
                                                bf16* __restrict__ A_bf,
                                                bf16* __restrict__ W1frag,
                                                bf16* __restrict__ Lfrag) {
  const int tid = threadIdx.x;
  if (tid < NV) {
    float s = 0.f;
    for (int j = 0; j < NV; ++j) s += fmaxf(A_param[tid * NV + j], 0.f);
    const float inv = 1.0f / fmaxf(s, 1e-6f);
    for (int j = 0; j < NV; ++j)
      A_bf[tid * NV + j] = (bf16)(fmaxf(A_param[tid * NV + j], 0.f) * inv);
  }
  // W1frag[((kt*4+nt)*32 + lane)*16 + i] = W1[o][d], o = nt*16+(lane&15),
  //   d = kt*32 + (lane>>4)*16 + i      (B-frag for T = H0 @ W1^T)
  for (int e = tid; e < 2 * 4 * 32 * 16; e += 256) {
    const int f    = e >> 9;
    const int lane = (e >> 4) & 31;
    const int i    = e & 15;
    const int kt   = f >> 2, nt = f & 3;
    const int o    = nt * 16 + (lane & 15);
    const int d    = kt * 32 + ((lane >> 4) << 4) + i;
    W1frag[e] = (bf16)W1[o * HID + d];
  }
  // Lfrag[((kt*4+nt)*32 + lane)*16 + i] = LB[K][u],
  //   K = kt*32 + (lane>>4)*16 + i,  u = nt*16 + (lane&15),  LB[(k,v)][u]=L_k[u][v]
  for (int e = tid; e < NKT * 4 * 32 * 16; e += 256) {
    const int f    = e >> 9;
    const int lane = (e >> 4) & 31;
    const int i    = e & 15;
    const int kt   = f >> 2, nt = f & 3;
    const int u    = nt * 16 + (lane & 15);
    const int K    = kt * 32 + ((lane >> 4) << 4) + i;
    const int kb   = K >> 6, v = K & 63;
    Lfrag[e] = (bf16)L_basis[((size_t)kb * NV + u) * NV + v];
  }
}

// ---- main fused kernel: 1 wave, 16 samples per block -----------------------
__global__ __launch_bounds__(32, 4) void gcn_main(
    const float* __restrict__ x,  const float* __restrict__ t_years,
    const float* __restrict__ alpha_w,
    const float* __restrict__ W0, const float* __restrict__ W2,
    const float* __restrict__ g0, const float* __restrict__ b0_,
    const float* __restrict__ g1, const float* __restrict__ b1_,
    const bf16* __restrict__ A_bf, const bf16* __restrict__ W1frag,
    const bf16* __restrict__ Lfrag, float* __restrict__ out) {
  __shared__ bf16 sAbf[NV * LDP];                  // adjacency
  __shared__ __align__(16) bf16 sW1f[8 * 32 * 16]; // W1^T, fragment-native
  __shared__ bf16  sXk[SPB * XLD];        // emb-scaled x, K=320 (dxdt A operand)
  __shared__ bf16  sH[NV * LDP];          // per-sample H0
  __shared__ bf16  sT[NV * LDP];          // per-sample T^T  ([o][j] !)
  __shared__ float sdx[SPB][NV];          // dxdt results
  __shared__ float su[NV];                // per-sample u = h1 . W2
  __shared__ float semb[SPB][NB];
  __shared__ float salpha[SPB];
  __shared__ float sW0[HID], sW2[HID], sg0[HID], sb0[HID], sg1[HID], sb1[HID];

  const int lane = threadIdx.x;
  const int half = lane >> 4;
  const int nl   = lane & 15;
  const int b0   = blockIdx.x * SPB;

  __builtin_prefetch(x + (size_t)b0 * NV + lane * 32, 0, 1);  // this block's x

  for (int e = lane; e < NV * NV; e += 32) {
    const int i = e >> 6, j = e & 63;
    sAbf[i * LDP + j] = A_bf[e];
  }
  for (int e = lane; e < (8 * 32 * 16) / 8; e += 32)   // 16B chunks
    ((uint4*)sW1f)[e] = ((const uint4*)W1frag)[e];
  for (int o = lane; o < HID; o += 32) {
    sW0[o] = W0[o]; sW2[o] = W2[o];
    sg0[o] = g0[o]; sb0[o] = b0_[o];
    sg1[o] = g1[o]; sb1[o] = b1_[o];
  }

  // time embedding + alpha; lanes 0..15 each own one sample (reconverges)
  {
    const int s = nl;
    const float t = t_years[b0 + s];
    const float w = 6.283185307179586f;
    const float c1 = cosf(w * t),        s1 = sinf(w * t);
    const float c2 = cosf(2.0f * w * t), s2 = sinf(2.0f * w * t);
    if (half == 0) {
      semb[s][0] = 1.0f; semb[s][1] = c1; semb[s][2] = s1;
      semb[s][3] = c2;   semb[s][4] = s2;
      salpha[s] = 1.0f / (1.0f + expf(-(alpha_w[0] + alpha_w[1] * c1 +
                                        alpha_w[2] * s1 + alpha_w[3] * c2 +
                                        alpha_w[4] * s2)));
    }
  }
  __syncthreads();

  // layer0 LN closed-form stats of relu(+/-W0) (uniform, once per block)
  float mp = 0.f, qp = 0.f, mm = 0.f, qm = 0.f;
  for (int o = 0; o < HID; ++o) {
    const float wp = fmaxf(sW0[o], 0.f), wn = fmaxf(-sW0[o], 0.f);
    mp += wp; qp = fmaf(wp, wp, qp);
    mm += wn; qm = fmaf(wn, wn, qm);
  }
  mp *= (1.0f / HID); qp *= (1.0f / HID);
  mm *= (1.0f / HID); qm *= (1.0f / HID);

  // Xk[s][k*64+v] = emb_k(s) * x[s][v]   (k=0 slice == x in bf16)
  for (int s = 0; s < SPB; ++s) {
    const float* xr = x + (size_t)(b0 + s) * NV;
    for (int K = lane; K < XK; K += 32)
      sXk[s * XLD + K] = (bf16)(semb[s][K >> 6] * xr[K & 63]);
  }

  // ---- dxdt GEMM: [16 x 320] x [320 x 64], all 16 M-rows live --------------
  {
    v8f acc[4] = {};
#pragma unroll 2
    for (int kt = 0; kt < NKT; ++kt) {
      const v16bf a = load_a_frag(sXk, XLD, 0, kt * 32, lane);
#pragma unroll
      for (int nt = 0; nt < 4; ++nt) {
        const v16bf bb = load_frag16(Lfrag + (size_t)((kt * 4 + nt) * 32 + lane) * 16);
        acc[nt] = wmma_bf16(a, bb, acc[nt]);
      }
    }
#pragma unroll
    for (int nt = 0; nt < 4; ++nt)
#pragma unroll
      for (int r = 0; r < 8; ++r)
        sdx[half * 8 + r][nt * 16 + nl] = acc[nt][r];
  }

  // ---- per-sample GNN ------------------------------------------------------
#pragma unroll 1
  for (int s = 0; s < SPB; ++s) {
    const bf16* xrow = sXk + s * XLD;   // bf16 x (emb_0 = 1 slice)

    // layer0 (collapsed): sc = A@x; H0 = LN(relu(sc_i * W0[o])) -> bf16
#pragma unroll
    for (int r = 0; r < 2; ++r) {
      const int i = lane + 32 * r;
      float sc = 0.f;
      for (int j = 0; j < NV; ++j)
        sc = fmaf((float)sAbf[i * LDP + j], (float)xrow[j], sc);
      const float as  = fabsf(sc);
      const float m_  = (sc < 0.f) ? mm : mp;
      const float q_  = (sc < 0.f) ? qm : qp;
      const float mu  = as * m_;
      const float var = sc * sc * (q_ - m_ * m_);
      const float rs  = rsqrtf(var + LN_EPS);
      for (int o = 0; o < HID; ++o) {
        const float e = fmaxf(sc * sW0[o], 0.f);
        sH[i * LDP + o] = (bf16)((e - mu) * rs * sg0[o] + sb0[o]);
      }
    }

    // GEMM1: T = H0 @ W1^T, stored TRANSPOSED (sT[o][j]) -> contiguous b128
#pragma unroll
    for (int mt = 0; mt < 4; ++mt) {
      const v16bf a0 = load_a_frag(sH, LDP, mt * 16, 0, lane);
      const v16bf a1 = load_a_frag(sH, LDP, mt * 16, 32, lane);
#pragma unroll
      for (int nt = 0; nt < 4; ++nt) {
        const v16bf w0f = load_frag16(sW1f + ((0 * 4 + nt) * 32 + lane) * 16);
        const v16bf w1f = load_frag16(sW1f + ((1 * 4 + nt) * 32 + lane) * 16);
        v8f acc = {};
        acc = wmma_bf16(a0, w0f, acc);
        acc = wmma_bf16(a1, w1f, acc);
        bf16* p = sT + (nt * 16 + nl) * LDP + mt * 16 + half * 8;
#pragma unroll
        for (int r = 0; r < 8; ++r) p[r] = (bf16)acc[r];   // 8 contiguous bf16
      }
    }

    // GEMM2: H1pre = A @ T; fused relu+LN+(.W2) on accumulators
#pragma unroll
    for (int mt = 0; mt < 4; ++mt) {
      const v16bf a0 = load_a_frag(sAbf, LDP, mt * 16, 0, lane);
      const v16bf a1 = load_a_frag(sAbf, LDP, mt * 16, 32, lane);
      v8f acc[4];
#pragma unroll
      for (int nt = 0; nt < 4; ++nt) {
        const v16bf bb0 = load_b_frag_T(sT, 0, nt * 16, lane);
        const v16bf bb1 = load_b_frag_T(sT, 32, nt * 16, lane);
        v8f c = {};
        c = wmma_bf16(a0, bb0, c);
        c = wmma_bf16(a1, bb1, c);
        acc[nt] = c;
      }
#pragma unroll
      for (int r = 0; r < 8; ++r) {
        float sum = 0.f, ss = 0.f;
#pragma unroll
        for (int nt = 0; nt < 4; ++nt) {
          const float e = fmaxf(acc[nt][r], 0.f);
          sum += e; ss = fmaf(e, e, ss);
        }
        sum = xor_reduce16(sum);
        ss  = xor_reduce16(ss);
        const float mu  = sum * (1.0f / HID);
        const float var = ss * (1.0f / HID) - mu * mu;
        const float rs  = rsqrtf(var + LN_EPS);
        float up = 0.f;
#pragma unroll
        for (int nt = 0; nt < 4; ++nt) {
          const int   o = nt * 16 + nl;
          const float e = fmaxf(acc[nt][r], 0.f);
          up = fmaf((e - mu) * rs * sg1[o] + sb1[o], sW2[o], up);
        }
        up = xor_reduce16(up);
        if (nl == 0) su[mt * 16 + half * 8 + r] = up;
      }
    }

    // layer2 aggregation g = A@u, blend with dxdt, store
    const float alpha = salpha[s];
#pragma unroll
    for (int r = 0; r < 2; ++r) {
      const int i = lane + 32 * r;
      float g = 0.f;
      for (int j = 0; j < NV; ++j)
        g = fmaf((float)sAbf[i * LDP + j], su[j], g);
      out[(size_t)(b0 + s) * NV + i] = sdx[s][i] + alpha * g;
    }
  }
}

// ---------------------------------------------------------------------------
extern "C" void kernel_launch(void* const* d_in, const int* in_sizes, int n_in,
                              void* d_out, int out_size, void* d_ws, size_t ws_size,
                              hipStream_t stream) {
  const float* x       = (const float*)d_in[0];
  const float* t_years = (const float*)d_in[1];
  const float* L_basis = (const float*)d_in[2];
  const float* A_param = (const float*)d_in[3];
  const float* alpha_w = (const float*)d_in[4];
  const float* W0      = (const float*)d_in[5];
  const float* W1      = (const float*)d_in[6];
  const float* W2      = (const float*)d_in[7];
  const float* ln0w    = (const float*)d_in[8];
  const float* ln0b    = (const float*)d_in[9];
  const float* ln1w    = (const float*)d_in[10];
  const float* ln1b    = (const float*)d_in[11];

  bf16* A_bf = (bf16*)d_ws;                   // 64*64 bf16 (8 KB)
  bf16* W1fr = A_bf + NV * NV;                // 8*32*16 bf16 (8 KB, frag-native)
  bf16* Lfr  = W1fr + 8 * 32 * 16;            // 40*32*16 bf16 (40 KB, frag-native)

  gcn_prep<<<1, 256, 0, stream>>>(A_param, W1, L_basis, A_bf, W1fr, Lfr);
  gcn_main<<<BLOCKS, 32, 0, stream>>>(x, t_years, alpha_w, W0, W2,
                                      ln0w, ln0b, ln1w, ln1b,
                                      A_bf, W1fr, Lfr, (float*)d_out);
}
